// LSTMCell_919123001533
// MI455X (gfx1250) — matrix operand order repaired and checked by reading, not compiled, not run on hardware
//
#include <hip/hip_runtime.h>

// LSTM, persistent-kernel formulation for MI455X (gfx1250).
// B=64, T=512, D_IN=D_HID=D_OUT=512.
//
// 32 blocks x 256 threads (8 wave32s). Block b owns hidden columns
// [16b, 16b+16). Its weight slice (1024 x 64 gate-cols) lives in LDS as bf16
// (128 KB). Cell state c lives in LDS (f32). Hidden state a is double-buffered
// in global workspace as bf16; one grid barrier per timestep.
//
// x is pre-converted to bf16 once (32 MB, L2-resident on a 192 MB L2) so the
// sequential loop reads A-fragments (x_t and a_{t-1}) straight from L2 with
// global_load_b128, and B-fragments from LDS with ds_load_b128.
// Fallback (small workspace): stage x_t into LDS per step.
//
// GEMM per step per block: M=64 (batch) x N=64 (4 gates x 16 cols) x K=1024,
// via v_wmma_f32_16x16x32_bf16. Wave w -> m-tile (w%4), n-tiles {2*(w/4),+1}.

typedef __attribute__((ext_vector_type(16))) __bf16 v16bf;
typedef __attribute__((ext_vector_type(8)))  __bf16 bf16x8;
typedef __attribute__((ext_vector_type(4)))  __bf16 bf16x4;
typedef __attribute__((ext_vector_type(8)))  float  v8f;

#define NBLK    32
#define NTHR    256
#define T_STEPS 512
#define BSZ     64
#define DHID    512
#define KDIM    1024   // D_IN + D_HID
#define XROW    (T_STEPS * 512)      // row stride of x / x_bf ([B][T][D])
#define XELEMS  (BSZ * T_STEPS * 512)

// LDS layout (dynamic):
//   [0)       W_lds : 64 cols x 1024 k, bf16, n-major     = 131072 B
//   [131072)  x_lds : 64 x 512 bf16 (fallback path only)  =  65536 B
//   [196608)  g_buf : 64 x 64 f32 (gate pre-activations)  =  16384 B
//   [212992)  c_lds : 64 x 16 f32 (cell state)            =   4096 B
#define SMEM_BYTES 217088

__device__ __forceinline__ v16bf cat8(bf16x8 lo, bf16x8 hi) {
  return __builtin_shufflevector(lo, hi, 0,1,2,3,4,5,6,7,8,9,10,11,12,13,14,15);
}

__device__ __forceinline__ float fast_sigmoid(float x) {
  return 1.0f / (1.0f + __expf(-x));
}
__device__ __forceinline__ float fast_tanh(float x) {
  x = fminf(fmaxf(x, -15.0f), 15.0f);
  float e = __expf(2.0f * x);
  return (e - 1.0f) / (e + 1.0f);
}

// One-time parallel fp32 -> bf16 conversion of the whole input sequence.
__global__ __launch_bounds__(NTHR) void cvt_x_bf16(const float* __restrict__ x,
                                                   __bf16* __restrict__ xbf) {
  const size_t stride = (size_t)gridDim.x * NTHR * 4;
  for (size_t i = ((size_t)blockIdx.x * NTHR + threadIdx.x) * 4; i < XELEMS;
       i += stride) {
    const float4 v = *(const float4*)(x + i);
    bf16x4 o = {(__bf16)v.x, (__bf16)v.y, (__bf16)v.z, (__bf16)v.w};
    *(bf16x4*)(xbf + i) = o;
  }
}

__global__ __launch_bounds__(NTHR) void lstm_persistent(
    const float* __restrict__ x, const __bf16* __restrict__ xbf,  // xbf==null -> stage in LDS
    const float* __restrict__ Wf, const float* __restrict__ Wi,
    const float* __restrict__ Wo, const float* __restrict__ Wc,
    const float* __restrict__ bf_, const float* __restrict__ bi_,
    const float* __restrict__ bo_, const float* __restrict__ bc_,
    __bf16* __restrict__ abuf0, __bf16* __restrict__ abuf1,
    float* __restrict__ out, unsigned int* __restrict__ ctr) {
  extern __shared__ char smem[];
  __bf16* W_lds = (__bf16*)smem;                     // [n][k] n-major, stride 1024
  __bf16* x_lds = (__bf16*)(smem + 131072);          // [b][d] stride 512 (fallback)
  float*  g_buf = (float*)(smem + 196608);           // [row][64]
  float*  c_lds = (float*)(smem + 212992);           // [row][16]

  const int tid  = threadIdx.x;
  const int blk  = blockIdx.x;
  const int lane = tid & 31;
  const int wave = tid >> 5;
  const int mt   = wave & 3;        // m-tile 0..3
  const int nt0  = (wave >> 2) * 2; // n-tiles {0,1} or {2,3}
  const int nt1  = nt0 + 1;
  const int l16  = lane & 15;
  const int half = lane >> 4;
  const bool use_xbf = (xbf != nullptr);

  // ---- one-time: load + convert this block's weight slice into LDS ----
  // local gate-col n = g*16 + hl  ->  W<g>[k*512 + blk*16 + hl]
  for (int i = tid; i < 64 * KDIM; i += NTHR) {
    int n = i >> 10, k = i & 1023;
    int g = n >> 4, hl = n & 15;
    const float* Wsrc = (g == 0) ? Wf : (g == 1) ? Wi : (g == 2) ? Wo : Wc;
    W_lds[n * KDIM + k] = (__bf16)Wsrc[k * DHID + blk * 16 + hl];
  }
  for (int i = tid; i < BSZ * 16; i += NTHR) c_lds[i] = 0.0f;
  __syncthreads();

  const int mrow = mt * 16 + l16;

  for (int t = 0; t < T_STEPS; ++t) {
    const __bf16* a_read  = (t & 1) ? abuf1 : abuf0;
    __bf16*       a_write = (t & 1) ? abuf0 : abuf1;

    if (!use_xbf) {
      // ---- fallback: stage x_t into LDS as bf16 (x is [B][T][D]) ----
      for (int i = tid * 4; i < BSZ * 512; i += NTHR * 4) {
        int b = i >> 9, d = i & 511;
        const float4 v = *(const float4*)(x + (size_t)b * XROW +
                                          (size_t)t * 512 + d);
        bf16x4 o = {(__bf16)v.x, (__bf16)v.y, (__bf16)v.z, (__bf16)v.w};
        *(bf16x4*)(x_lds + i) = o;
        if (t + 1 < T_STEPS)
          __builtin_prefetch(x + (size_t)b * XROW + (size_t)(t + 1) * 512 + d, 0, 0);
      }
      __syncthreads();
    }

    // ---- GEMM: g = [x_t, a_{t-1}] @ Wslice ----
    v8f acc0 = {}, acc1 = {};
    const __bf16* bcol0 = W_lds + (nt0 * 16 + l16) * KDIM + half * 8;
    const __bf16* bcol1 = W_lds + (nt1 * 16 + l16) * KDIM + half * 8;

    // K = 0..511 : A from x_t
    if (use_xbf) {
      const __bf16* xrow = xbf + (size_t)mrow * XROW + (size_t)t * 512 + half * 8;
      #pragma unroll 4
      for (int kt = 0; kt < 16; ++kt) {
        const int k0 = kt * 32;
        v16bf afrag = cat8(*(const bf16x8*)(xrow + k0), *(const bf16x8*)(xrow + k0 + 16));
        v16bf b0 = cat8(*(const bf16x8*)(bcol0 + k0), *(const bf16x8*)(bcol0 + k0 + 16));
        v16bf b1 = cat8(*(const bf16x8*)(bcol1 + k0), *(const bf16x8*)(bcol1 + k0 + 16));
        acc0 = __builtin_amdgcn_wmma_f32_16x16x32_bf16(false, afrag, false, b0,
                                                       (short)0, acc0, false, false);
        acc1 = __builtin_amdgcn_wmma_f32_16x16x32_bf16(false, afrag, false, b1,
                                                       (short)0, acc1, false, false);
      }
    } else {
      const __bf16* xrow = x_lds + mrow * 512 + half * 8;
      #pragma unroll 4
      for (int kt = 0; kt < 16; ++kt) {
        const int k0 = kt * 32;
        v16bf afrag = cat8(*(const bf16x8*)(xrow + k0), *(const bf16x8*)(xrow + k0 + 16));
        v16bf b0 = cat8(*(const bf16x8*)(bcol0 + k0), *(const bf16x8*)(bcol0 + k0 + 16));
        v16bf b1 = cat8(*(const bf16x8*)(bcol1 + k0), *(const bf16x8*)(bcol1 + k0 + 16));
        acc0 = __builtin_amdgcn_wmma_f32_16x16x32_bf16(false, afrag, false, b0,
                                                       (short)0, acc0, false, false);
        acc1 = __builtin_amdgcn_wmma_f32_16x16x32_bf16(false, afrag, false, b1,
                                                       (short)0, acc1, false, false);
      }
    }
    // K = 512..1023 : A from a_{t-1} (global bf16, L2-resident)
    {
      const __bf16* arow = a_read + mrow * DHID + half * 8;
      #pragma unroll 4
      for (int kt = 16; kt < 32; ++kt) {
        const int k0 = kt * 32;
        const int ka = k0 - 512;
        v16bf afrag = cat8(*(const bf16x8*)(arow + ka), *(const bf16x8*)(arow + ka + 16));
        v16bf b0 = cat8(*(const bf16x8*)(bcol0 + k0), *(const bf16x8*)(bcol0 + k0 + 16));
        v16bf b1 = cat8(*(const bf16x8*)(bcol1 + k0), *(const bf16x8*)(bcol1 + k0 + 16));
        acc0 = __builtin_amdgcn_wmma_f32_16x16x32_bf16(false, afrag, false, b0,
                                                       (short)0, acc0, false, false);
        acc1 = __builtin_amdgcn_wmma_f32_16x16x32_bf16(false, afrag, false, b1,
                                                       (short)0, acc1, false, false);
      }
    }

    // ---- scatter C tiles to g_buf (C layout: M = v + 8*half, N = lane%16) ----
    #pragma unroll
    for (int v = 0; v < 8; ++v) {
      int r = mt * 16 + half * 8 + v;
      g_buf[r * 64 + nt0 * 16 + l16] = acc0[v];
      g_buf[r * 64 + nt1 * 16 + l16] = acc1[v];
    }
    __syncthreads();

    // ---- gates + state update for this block's 16 hidden cols ----
    for (int i = tid; i < BSZ * 16; i += NTHR) {
      int r = i >> 4, hl = i & 15;
      int gc = blk * 16 + hl;
      float gf = g_buf[r * 64 +      hl] + bf_[gc];
      float gi = g_buf[r * 64 + 16 + hl] + bi_[gc];
      float go = g_buf[r * 64 + 32 + hl] + bo_[gc];
      float gg = g_buf[r * 64 + 48 + hl] + bc_[gc];
      float f  = fast_sigmoid(gf);
      float ii = fast_sigmoid(gi);
      float o  = fast_sigmoid(go);
      float cn = f * c_lds[i] + ii * fast_tanh(gg);
      c_lds[i] = cn;
      float an = o * fast_tanh(cn);
      a_write[r * DHID + gc] = (__bf16)an;
      if (t == T_STEPS - 1) out[r * DHID + gc] = an;
    }

    // ---- grid-wide barrier (monotone counter, one arrival per block) ----
    __syncthreads();
    if (tid == 0) {
      __threadfence();
      __hip_atomic_fetch_add(ctr, 1u, __ATOMIC_ACQ_REL, __HIP_MEMORY_SCOPE_AGENT);
      const unsigned int target = (unsigned int)(t + 1) * NBLK;
      while (__hip_atomic_load(ctr, __ATOMIC_ACQUIRE, __HIP_MEMORY_SCOPE_AGENT) < target)
        __builtin_amdgcn_s_sleep(1);
    }
    __syncthreads();
    __threadfence();  // acquire-side: make peers' a_write visible to all lanes
  }
}

extern "C" void kernel_launch(void* const* d_in, const int* in_sizes, int n_in,
                              void* d_out, int out_size, void* d_ws, size_t ws_size,
                              hipStream_t stream) {
  const float* x   = (const float*)d_in[0];
  const float* Wf  = (const float*)d_in[1];
  const float* Wc  = (const float*)d_in[2];
  const float* Wi  = (const float*)d_in[3];
  const float* Wo  = (const float*)d_in[4];
  const float* bf_ = (const float*)d_in[5];
  const float* bc_ = (const float*)d_in[6];
  const float* bi_ = (const float*)d_in[7];
  const float* bo_ = (const float*)d_in[8];
  float* out = (float*)d_out;

  // Workspace layout: [ctr 256B][abuf0 64KB][abuf1 64KB][x_bf 32MB (optional)]
  const size_t a_off0 = 256;
  const size_t a_off1 = 256 + (size_t)BSZ * DHID * 2;
  const size_t x_off  = 256 + 2 * (size_t)BSZ * DHID * 2;
  const size_t need   = x_off + (size_t)XELEMS * 2;

  unsigned int* ctr = (unsigned int*)d_ws;
  __bf16* abuf0 = (__bf16*)((char*)d_ws + a_off0);
  __bf16* abuf1 = (__bf16*)((char*)d_ws + a_off1);
  __bf16* xbf   = (ws_size >= need) ? (__bf16*)((char*)d_ws + x_off) : nullptr;

  // zero barrier counter + both a-state buffers (a0 = 0); capture-safe
  hipMemsetAsync(d_ws, 0, x_off, stream);

  if (xbf) cvt_x_bf16<<<2048, NTHR, 0, stream>>>(x, xbf);

  hipFuncSetAttribute((const void*)lstm_persistent,
                      hipFuncAttributeMaxDynamicSharedMemorySize, SMEM_BYTES);

  lstm_persistent<<<NBLK, NTHR, SMEM_BYTES, stream>>>(
      x, xbf, Wf, Wi, Wo, Wc, bf_, bi_, bo_, bc_, abuf0, abuf1, out, ctr);
}
